// GAT_layer_21629455303171
// MI455X (gfx1250) — matrix operand order repaired and checked by reading, not compiled, not run on hardware
//
#include <hip/hip_runtime.h>
#include <hip/hip_bf16.h>
#include <stdint.h>

typedef __attribute__((ext_vector_type(16))) _Float16 v16h;
typedef __attribute__((ext_vector_type(8)))  float    v8f;
typedef __attribute__((ext_vector_type(4)))  unsigned int u32x4;
typedef __attribute__((ext_vector_type(4)))  int          i32x4;
typedef __attribute__((ext_vector_type(4)))  float        f32x4;

#define Nn 8192
#define Hh 128

// ---------------------------------------------------------------------------
// Kernel A: Wh = node @ W^T + b (f32), Whi = Wh . a1, Whj = Wh . a2
// One block per row; thread h computes one output feature. W (64KB) is
// L2-resident; node row staged in LDS.
// ---------------------------------------------------------------------------
__global__ void __launch_bounds__(128)
wh_kernel(const float* __restrict__ node, const float* __restrict__ W,
          const float* __restrict__ b, const float* __restrict__ a1,
          const float* __restrict__ a2, float* __restrict__ Wh,
          float* __restrict__ Whi, float* __restrict__ Whj) {
  __shared__ float row[Hh];
  __shared__ float p1[Hh];
  __shared__ float p2[Hh];
  const int i = blockIdx.x;
  const int h = threadIdx.x;
  row[h] = node[(size_t)i * Hh + h];
  __syncthreads();
  float acc = b[h];
  const float* wrow = W + (size_t)h * Hh;
#pragma unroll 8
  for (int k = 0; k < Hh; ++k) acc += row[k] * wrow[k];
  Wh[(size_t)i * Hh + h] = acc;
  p1[h] = acc * a1[h];
  p2[h] = acc * a2[h];
  __syncthreads();
#pragma unroll
  for (int s = 64; s > 0; s >>= 1) {
    if (h < s) { p1[h] += p1[h + s]; p2[h] += p2[h + s]; }
    __syncthreads();
  }
  if (h == 0) { Whi[i] = p1[0]; Whj[i] = p2[0]; }
}

// ---------------------------------------------------------------------------
// Kernel B: WhT[h][i] = (f16) Wh[i][h]  — LDS-tiled transpose so the main
// kernel's WMMA B-operand reads are contiguous (32B per lane).
// ---------------------------------------------------------------------------
__global__ void __launch_bounds__(256)
transpose_kernel(const float* __restrict__ Wh, _Float16* __restrict__ WhT) {
  __shared__ _Float16 tile[64][64 + 8];
  const int i0 = blockIdx.x * 64;
  const int h0 = blockIdx.y * 64;
  const int t = threadIdx.x;
#pragma unroll
  for (int rep = 0; rep < 16; ++rep) {
    int idx = rep * 256 + t;
    int ii = idx >> 6, hh = idx & 63;
    tile[hh][ii] = (_Float16)Wh[(size_t)(i0 + ii) * Hh + h0 + hh];
  }
  __syncthreads();
#pragma unroll
  for (int rep = 0; rep < 16; ++rep) {
    int idx = rep * 256 + t;
    int hh = idx >> 6, ii = idx & 63;
    WhT[(size_t)(h0 + hh) * Nn + i0 + ii] = tile[hh][ii];
  }
}

// ---------------------------------------------------------------------------
// Kernel C: fused masked-softmax attention * Wh + Wh (flash-attention style).
// Workgroup = 16-row i-tile, 8 wave32; wave w owns h-slice [16w, 16w+16).
// Per 128-wide j-chunk: cooperatively build P (f16) in a double-buffered LDS
// tile (one barrier per chunk), then 4x v_wmma_f32_16x16x32_f16 per wave
// (A from LDS, B = WhT f16 from global/L2). Branchless mask+exp: v_max_f32
// for leaky-relu, unconditional v_exp_f32, v_cndmask*mul for the adj mask.
// Row denominators accumulated per-thread, reduced in LDS, applied at end.
// ---------------------------------------------------------------------------
__global__ void __launch_bounds__(256)
gat_attn_kernel(const int* __restrict__ adj, const float* __restrict__ Whi,
                const float* __restrict__ Whj, const float* __restrict__ Wh,
                const _Float16* __restrict__ WhT, float* __restrict__ out) {
  __shared__ _Float16 P[2][16][136];  // double-buffered, +8 f16 pad per row
  __shared__ float dred[16][16];
  __shared__ float dfin[16];

  const int tid = threadIdx.x;
  const int i0 = blockIdx.x * 16;
  const int r = tid >> 4;          // row in i-tile, 0..15
  const int c16 = tid & 15;        // 8-wide j sub-slice owner
  const int lane = tid & 31;       // wave32 lane
  const int wave = tid >> 5;       // 0..7
  const int h0 = wave * 16;

  const float whi_r = Whi[i0 + r];
  const int* adjrow = adj + (size_t)(i0 + r) * Nn;

  union U16 { v16h v; _Float16 h[16]; u32x4 q[2]; };

  v8f c = {};
  float dloc = 0.f;

  // WMMA operand addressing (per the CDNA5 16-bit A / B VGPR layouts)
  const int aoff0 = (lane < 16) ? 0 : 8;    // A: lanes>=16 hold K 8..15 / 24..31
  const int boff  = (lane < 16) ? 0 : 8;    // B: lanes>=16 hold K 8..15 / 24..31
  const int arow  = lane & 15;
  const _Float16* Brow = WhT + (size_t)(h0 + (lane & 15)) * Nn;

  int buf = 0;
  for (int j0 = 0; j0 < Nn; j0 += 128, buf ^= 1) {
    // ---- phase 1: cooperative P tile (16x128), branchless ----
    const int jb = j0 + c16 * 8;
    i32x4 a0 = *(const i32x4*)(adjrow + jb);
    i32x4 a1v = *(const i32x4*)(adjrow + jb + 4);
    f32x4 w0 = *(const f32x4*)(Whj + jb);
    f32x4 w1 = *(const f32x4*)(Whj + jb + 4);
    if (j0 + 128 < Nn)
      __builtin_prefetch(adjrow + jb + 128, 0, 1);  // global_prefetch_b8

    float wj[8] = {w0[0], w0[1], w0[2], w0[3], w1[0], w1[1], w1[2], w1[3]};
    int   av[8] = {a0[0], a0[1], a0[2], a0[3], a1v[0], a1v[1], a1v[2], a1v[3]};
#pragma unroll
    for (int u = 0; u < 8; ++u) {
      float x = whi_r + wj[u];
      float s = fmaxf(x, 0.2f * x);              // leaky_relu(0.2) = max(x,0.2x)
      float e = __expf(s);                       // unconditional v_exp_f32
      float m = (av[u] > 0) ? 1.0f : 0.0f;       // v_cndmask, no branch
      float p = e * m;                           // masked -> exact 0
      dloc += p;
      P[buf][r][c16 * 8 + u] = (_Float16)p;
    }
    __syncthreads();

    // ---- phase 2: 4 K-steps of WMMA per wave ----
#pragma unroll
    for (int kk = 0; kk < 4; ++kk) {
      U16 A, B;
      A.q[0] = *(const u32x4*)&P[buf][arow][kk * 32 + aoff0];       // K 0..7 / 8..15
      A.q[1] = *(const u32x4*)&P[buf][arow][kk * 32 + aoff0 + 16];  // K 16..23 / 24..31
      const _Float16* bp = Brow + j0 + kk * 32 + boff;
      B.q[0] = *(const u32x4*)(bp);
      B.q[1] = *(const u32x4*)(bp + 16);
      c = __builtin_amdgcn_wmma_f32_16x16x32_f16(false, A.v, false, B.v,
                                                 (short)0, c, false, false);
    }
    // no second barrier: double buffering makes next write safe after the
    // barrier of the following iteration
  }

  // ---- denominator reduction ----
  dred[r][c16] = dloc;
  __syncthreads();
  if (tid < 16) {
    float s = 0.f;
#pragma unroll
    for (int u = 0; u < 16; ++u) s += dred[tid][u];
    dfin[tid] = s;
  }
  __syncthreads();

  // ---- epilogue: scale by 1/rowsum (v_rcp_f32), add residual, store ----
  const int mbase = (lane < 16) ? 0 : 8;
  const int n = lane & 15;
#pragma unroll
  for (int v = 0; v < 8; ++v) {
    int M = v + mbase;                          // C layout: VGPR v <-> M=v / v+8
    size_t idx = (size_t)(i0 + M) * Hh + h0 + n;
    float inv = __builtin_amdgcn_rcpf(dfin[M]);
    out[idx] = c[v] * inv + Wh[idx];
  }
}

// ---------------------------------------------------------------------------
extern "C" void kernel_launch(void* const* d_in, const int* in_sizes, int n_in,
                              void* d_out, int out_size, void* d_ws, size_t ws_size,
                              hipStream_t stream) {
  const float* node = (const float*)d_in[0];   // [8192,128]
  const float* edge = (const float*)d_in[1];   // [8192,8192,1]
  const int*   adj  = (const int*)d_in[2];     // [8192,8192]
  const float* W_w  = (const float*)d_in[3];   // [128,128]
  const float* W_b  = (const float*)d_in[4];   // [128]
  const float* a1_w = (const float*)d_in[5];   // [1,128]
  const float* a2_w = (const float*)d_in[6];   // [1,128]

  float* out_node = (float*)d_out;                       // [8192,128]
  float* out_edge = out_node + (size_t)Nn * Hh;          // [8192,8192]

  // workspace layout
  float*    Wh  = (float*)d_ws;                                       // 4 MB
  _Float16* WhT = (_Float16*)((char*)d_ws + (size_t)4 * 1024 * 1024); // 2 MB
  float*    Whi = (float*)((char*)d_ws + (size_t)6 * 1024 * 1024);    // 32 KB
  float*    Whj = Whi + Nn;                                           // 32 KB

  wh_kernel<<<Nn, 128, 0, stream>>>(node, W_w, W_b, a1_w, a2_w, Wh, Whi, Whj);
  transpose_kernel<<<dim3(Nn / 64, Hh / 64), 256, 0, stream>>>(Wh, WhT);
  gat_attn_kernel<<<Nn / 16, 256, 0, stream>>>(adj, Whi, Whj, Wh, WhT, out_node);

  // edge_fea pass-through (256 MB D2D at HBM bandwidth, capture-safe)
  hipMemcpyAsync(out_edge, edge, (size_t)Nn * Nn * sizeof(float),
                 hipMemcpyDeviceToDevice, stream);
}
